// BMamba_mini_44925357916653
// MI455X (gfx1250) — matrix operand order repaired and checked by reading, not compiled, not run on hardware
//
#include <hip/hip_runtime.h>
#include <hip/hip_bf16.h>
#include <math.h>

typedef __attribute__((ext_vector_type(16))) _Float16 v16h;
typedef __attribute__((ext_vector_type(8)))  float    v8f;

// ---------------- canonical parameter table ----------------
#define TI_X     0
#define TI_S11   1   // +0 w, +1 b, +2 gamma, +3 beta, +4 mean, +5 var
#define TI_S12   7
#define TI_S21   13
#define TI_S22   19
#define TI_S3    25
#define TI_LASTW 31
#define TI_LASTB 32
#define TI_BLK   33  // 17 per block: n1g,n1b,mng,mnb,inproj,convw,convb,xproj,dtw,dtb,Alog,Dp,outproj,fc1w,fc1b,fc2w,fc2b
#define NTAB     441

struct PTab { const float* p[NTAB]; };

// ---------------- workspace float offsets ----------------
static constexpr size_t SS_OFF  = 0ull;        // 640*4*32*32
static constexpr size_t SD_OFF  = 2621440ull;
static constexpr size_t Y8_OFF  = 5242880ull;  // 640*8*16*16
static constexpr size_t T_OFF   = 6553600ull;  // 320*32 tokens
static constexpr size_t H_OFF   = 6563840ull;
static constexpr size_t U_OFF   = 6574080ull;
static constexpr size_t XZ_OFF  = 6584320ull;  // 320*128
static constexpr size_t XC_OFF  = 6625280ull;  // 320*64
static constexpr size_t DBC_OFF = 6645760ull;  // 320*48
static constexpr size_t DT_OFF  = 6661120ull;  // 320*64
static constexpr size_t YS_OFF  = 6681600ull;  // 320*64
static constexpr size_t FF_OFF  = 6702080ull;  // 320*64

__device__ __forceinline__ float sigm_(float x)  { return 1.0f / (1.0f + __expf(-x)); }
__device__ __forceinline__ float silu_(float x)  { return x * sigm_(x); }
__device__ __forceinline__ float splus_(float x) { return x > 20.0f ? x : log1pf(__expf(x)); }
__device__ __forceinline__ float gelu_(float x)  { return 0.5f * x * (1.0f + erff(x * 0.70710678118654752f)); }

// Async global->LDS path (gfx1250 GLOBAL_LOAD_ASYNC_TO_LDS_B32, ASYNCcnt).
// Probe result: builtin exists and takes int* pointer params.
#if defined(__gfx1250__) && __has_builtin(__builtin_amdgcn_global_load_async_to_lds_b32) && \
    __has_builtin(__builtin_amdgcn_s_wait_asynccnt)
#define USE_ASYNC_LDS 1
#define AS1CAST(p) ((__attribute__((address_space(1))) int*)(void*)(p))
#define AS3CAST(p) ((__attribute__((address_space(3))) int*)(void*)(p))
#else
#define USE_ASYNC_LDS 0
#endif

// =====================================================================
// Stage 1: temporal diffs + stem11 / stem12 (conv7x7 s2 + BN + ReLU + pool2)
// grid: 640 frames * 4 quadrants, 256 threads; each thread -> one pooled px
// =====================================================================
__global__ void stem1_kernel(PTab tb, float* __restrict__ wsf) {
    const float* __restrict__ x   = tb.p[TI_X];
    const float* w11 = tb.p[TI_S11+0]; const float* b11 = tb.p[TI_S11+1];
    const float* g11 = tb.p[TI_S11+2]; const float* be11 = tb.p[TI_S11+3];
    const float* mu11 = tb.p[TI_S11+4]; const float* va11 = tb.p[TI_S11+5];
    const float* w12 = tb.p[TI_S12+0]; const float* b12 = tb.p[TI_S12+1];
    const float* g12 = tb.p[TI_S12+2]; const float* be12 = tb.p[TI_S12+3];
    const float* mu12 = tb.p[TI_S12+4]; const float* va12 = tb.p[TI_S12+5];
    float* ss = wsf + SS_OFF;
    float* sd = wsf + SD_OFF;

    __shared__ float sw11[196];
    __shared__ float sw12[784];
    for (int i = threadIdx.x; i < 196; i += 256) sw11[i] = w11[i];
    for (int i = threadIdx.x; i < 784; i += 256) sw12[i] = w12[i];
    __syncthreads();

    int fq = blockIdx.x;
    int f  = fq >> 2;
    int q  = fq & 3;
    int b  = f / 320;
    int d  = f - b * 320;
    int p  = q * 256 + threadIdx.x;
    int py = p >> 5, px = p & 31;

    const float* base = x + (size_t)f * 16384;
    int dm2 = d - 2 < 0 ? 0 : d - 2;
    int dm1 = d - 1 < 0 ? 0 : d - 1;
    int dp1 = d + 1 > 319 ? 319 : d + 1;
    int dp2 = d + 2 > 319 ? 319 : d + 2;
    const float* pbm2 = x + ((size_t)b * 320 + dm2) * 16384;
    const float* pbm1 = x + ((size_t)b * 320 + dm1) * 16384;
    const float* pbp1 = x + ((size_t)b * 320 + dp1) * 16384;
    const float* pbp2 = x + ((size_t)b * 320 + dp2) * 16384;

    float po11[4], po12[4];
    #pragma unroll
    for (int oc = 0; oc < 4; ++oc) { po11[oc] = -1e30f; po12[oc] = -1e30f; }

    for (int sy = 0; sy < 2; ++sy)
    for (int sx = 0; sx < 2; ++sx) {
        int cy = 2 * py + sy, cx = 2 * px + sx;
        float a11[4], a12[4];
        #pragma unroll
        for (int oc = 0; oc < 4; ++oc) { a11[oc] = b11[oc]; a12[oc] = b12[oc]; }
        for (int ky = 0; ky < 7; ++ky) {
            int iy = 2 * cy - 3 + ky;
            bool yok = (unsigned)iy < 128u;
            #pragma unroll
            for (int kx = 0; kx < 7; ++kx) {
                int ix = 2 * cx - 3 + kx;
                bool ok = yok && ((unsigned)ix < 128u);
                int off = iy * 128 + ix;
                float v0  = ok ? base[off] : 0.0f;
                float vm2 = ok ? pbm2[off] : 0.0f;
                float vm1 = ok ? pbm1[off] : 0.0f;
                float vp1 = ok ? pbp1[off] : 0.0f;
                float vp2 = ok ? pbp2[off] : 0.0f;
                float c0 = vm1 - vm2, c1 = v0 - vm1, c2 = vp1 - v0, c3 = vp2 - vp1;
                int kk = ky * 7 + kx;
                #pragma unroll
                for (int oc = 0; oc < 4; ++oc) {
                    a11[oc] += sw11[oc * 49 + kk] * v0;
                    const float* wr = &sw12[oc * 196 + kk];
                    a12[oc] += wr[0] * c0 + wr[49] * c1 + wr[98] * c2 + wr[147] * c3;
                }
            }
        }
        #pragma unroll
        for (int oc = 0; oc < 4; ++oc) {
            float v = (a11[oc] - mu11[oc]) * rsqrtf(va11[oc] + 1e-5f) * g11[oc] + be11[oc];
            v = v > 0.0f ? v : 0.0f;
            po11[oc] = fmaxf(po11[oc], v);
            float w = (a12[oc] - mu12[oc]) * rsqrtf(va12[oc] + 1e-5f) * g12[oc] + be12[oc];
            w = w > 0.0f ? w : 0.0f;
            po12[oc] = fmaxf(po12[oc], w);
        }
    }
    #pragma unroll
    for (int oc = 0; oc < 4; ++oc) {
        size_t o = ((size_t)f * 4 + oc) * 1024 + py * 32 + px;
        ss[o] = po11[oc];
        sd[o] = po12[oc];
    }
}

// =====================================================================
// Stage 2 (WMMA implicit GEMM): stem21(0.5ss+0.5sd) & stem22(sd),
// conv7x7 s1 pad3 + BN + ReLU + pool2 -> y8.
// A gathers are UNCONDITIONAL: padded k's map to LDS offset 0 and their
// product is annihilated by the zero-padded B (weights) fragment.
// Convs are processed sequentially to halve live accumulator pressure.
// =====================================================================
__device__ __forceinline__ v8f conv_tile(const float* __restrict__ inp,
                                         const _Float16* __restrict__ wrow,
                                         const int* __restrict__ lut,
                                         int pbase, int akb, int bkb) {
    v8f acc = {};
    for (int k0 = 0; k0 < 224; k0 += 32) {
        v16h a, b;
        #pragma unroll
        for (int e = 0; e < 8; ++e) {
            a[e]     = (_Float16)inp[pbase + lut[k0 + akb + e]];
            a[e + 8] = (_Float16)inp[pbase + lut[k0 + 16 + akb + e]];
        }
        #pragma unroll
        for (int e = 0; e < 16; ++e) b[e] = wrow[k0 + bkb + (e & 7) + ((e >> 3) << 4)];
        acc = __builtin_amdgcn_wmma_f32_16x16x32_f16(false, a, false, b, (short)0, acc,
                                                     false, false);
    }
    return acc;
}

__global__ void stem2_kernel(PTab tb, float* __restrict__ wsf) {
    const float* w21 = tb.p[TI_S21+0]; const float* bi21 = tb.p[TI_S21+1];
    const float* g21 = tb.p[TI_S21+2]; const float* be21 = tb.p[TI_S21+3];
    const float* mu21 = tb.p[TI_S21+4]; const float* va21 = tb.p[TI_S21+5];
    const float* w22 = tb.p[TI_S22+0]; const float* bi22 = tb.p[TI_S22+1];
    const float* g22 = tb.p[TI_S22+2]; const float* be22 = tb.p[TI_S22+3];
    const float* mu22 = tb.p[TI_S22+4]; const float* va22 = tb.p[TI_S22+5];

    __shared__ float    in1[4][38][38];      // 0.5*(ss+sd), zero halo
    __shared__ float    in2[4][38][38];      // sd, zero halo
    __shared__ _Float16 swh[2 * 16 * 224];   // zero-padded weights [conv][16][224]
    __shared__ int      lut[224];            // k -> ic*1444 + ky*38 + kx (0 for pad k)

    int f = blockIdx.x;
    const float* ss = wsf + SS_OFF + (size_t)f * 4096;
    const float* sd = wsf + SD_OFF + (size_t)f * 4096;
    float* y8 = wsf + Y8_OFF;

    // weight tile (f16, padded) + im2col LUT
    for (int i = threadIdx.x; i < 2 * 16 * 224; i += 256) {
        int c = i / 3584;
        int r = i - c * 3584;
        int oc = r / 224;
        int k  = r - oc * 224;
        const float* w = c ? w22 : w21;
        swh[i] = (oc < 8 && k < 196) ? (_Float16)w[oc * 196 + k] : (_Float16)0.0f;
    }
    for (int k = threadIdx.x; k < 224; k += 256) {
        if (k < 196) {
            int ic = k / 49;
            int rr = k - ic * 49;
            int ky = rr / 7;
            int kx = rr - ky * 7;
            lut[k] = ic * 1444 + ky * 38 + kx;
        } else {
            lut[k] = 0;   // safe dummy; B fragment is zero there
        }
    }

#if USE_ASYNC_LDS
    // zero the halo (async copies never touch it -> no DS/async overlap)
    for (int i = threadIdx.x; i < 5776; i += 256) {
        int c = i / 1444;
        int r = i - c * 1444;
        int yy = r / 38, xx = r - (r / 38) * 38;
        bool interior = (yy >= 3) && (yy < 35) && (xx >= 3) && (xx < 35);
        if (!interior) { (&in1[0][0][0])[i] = 0.0f; (&in2[0][0][0])[i] = 0.0f; }
    }
    // async-copy the 4x32x32 interiors: ss -> in1, sd -> in2
    for (int idx = threadIdx.x; idx < 4096; idx += 256) {
        int c = idx >> 10;
        int r = idx & 1023;
        int iy = r >> 5, ix = r & 31;
        __builtin_amdgcn_global_load_async_to_lds_b32(
            AS1CAST(ss + c * 1024 + r), AS3CAST(&in1[c][iy + 3][ix + 3]), 0, 0);
        __builtin_amdgcn_global_load_async_to_lds_b32(
            AS1CAST(sd + c * 1024 + r), AS3CAST(&in2[c][iy + 3][ix + 3]), 0, 0);
    }
    __builtin_amdgcn_s_wait_asynccnt(0);
    __syncthreads();
    // in1 currently holds ss; fold in sd: in1 = 0.5*(ss+sd) (halo stays 0)
    {
        float* f1 = &in1[0][0][0];
        const float* f2 = &in2[0][0][0];
        for (int i = threadIdx.x; i < 5776; i += 256) f1[i] = 0.5f * (f1[i] + f2[i]);
    }
    __syncthreads();
#else
    for (int i = threadIdx.x; i < 5776; i += 256) {
        int c  = i / 1444;
        int r  = i - c * 1444;
        int yy = r / 38, xx = r - (r / 38) * 38;
        int iy = yy - 3, ix = xx - 3;
        bool ok = ((unsigned)iy < 32u) && ((unsigned)ix < 32u);
        float vss = ok ? ss[c * 1024 + iy * 32 + ix] : 0.0f;
        float vsd = ok ? sd[c * 1024 + iy * 32 + ix] : 0.0f;
        in1[c][yy][xx] = 0.5f * (vss + vsd);
        in2[c][yy][xx] = vsd;
    }
    __syncthreads();
#endif

    int lane = threadIdx.x & 31;
    int wave = threadIdx.x >> 5;
    int akb = (lane < 16) ? 0 : 8;
    int bkb = (lane < 16) ? 0 : 8;
    int oc  = lane & 15;

    // hoisted BN affine (guarded reads; lanes oc>=8 never store)
    float scv[2] = {0.0f, 0.0f}, shv[2] = {0.0f, 0.0f};
    if (oc < 8) {
        scv[0] = g21[oc] * rsqrtf(va21[oc] + 1e-5f);
        shv[0] = be21[oc] + (bi21[oc] - mu21[oc]) * scv[0];
        scv[1] = g22[oc] * rsqrtf(va22[oc] + 1e-5f);
        shv[1] = be22[oc] + (bi22[oc] - mu22[oc]) * scv[1];
    }

    // 32 row-pair tiles per frame, 8 waves -> 4 each; EXEC uniform at WMMA
    for (int tp = wave; tp < 32; tp += 8) {
        int py  = tp >> 1;            // pooled row
        int cx0 = (tp & 1) * 16;      // conv column base of the 16-pixel tile
        int px  = cx0 + (lane & 15);  // this lane's conv column (A row)
        int pbE = (2 * py) * 38 + px;
        int pbO = (2 * py + 1) * 38 + px;

        float qv[2][4];
        #pragma unroll
        for (int c = 0; c < 2; ++c) {
            const float*    inp = c ? &in2[0][0][0] : &in1[0][0][0];
            const _Float16* wr  = &swh[c * 3584 + oc * 224];
            v8f ev = conv_tile(inp, wr, lut, pbE, akb, bkb);
            v8f ov = conv_tile(inp, wr, lut, pbO, akb, bkb);
            float sc = scv[c], sh = shv[c];
            #pragma unroll
            for (int j = 0; j < 4; ++j) {
                float a0 = fmaxf(ev[2 * j]     * sc + sh, 0.0f);
                float a1 = fmaxf(ev[2 * j + 1] * sc + sh, 0.0f);
                float a2 = fmaxf(ov[2 * j]     * sc + sh, 0.0f);
                float a3 = fmaxf(ov[2 * j + 1] * sc + sh, 0.0f);
                qv[c][j] = fmaxf(fmaxf(a0, a1), fmaxf(a2, a3));
            }
        }
        if (oc < 8) {
            int halfp = (lane < 16) ? 0 : 4;   // pooled-col offset of this lane half
            #pragma unroll
            for (int j = 0; j < 4; ++j) {
                int pxcol = (cx0 >> 1) + halfp + j;
                y8[((size_t)f * 8 + oc) * 256 + py * 16 + pxcol] = 0.5f * (qv[0][j] + qv[1][j]);
            }
        }
    }
}

// =====================================================================
// Stage 3: 3D conv (8->32, 2x5x5, stride 2,1,1, pad 0,2,2) + BN + sigmoid
// mask-pool -> tokens T[b][od][oc]. grid = 2*32*160 blocks, 256 threads.
// =====================================================================
__global__ void stem3_kernel(PTab tb, float* __restrict__ wsf) {
    const float* w3 = tb.p[TI_S3+0]; const float* b3 = tb.p[TI_S3+1];
    const float* g3 = tb.p[TI_S3+2]; const float* be3 = tb.p[TI_S3+3];
    const float* mu3 = tb.p[TI_S3+4]; const float* va3 = tb.p[TI_S3+5];
    const float* y8 = wsf + Y8_OFF;
    float* T = wsf + T_OFF;

    int id = blockIdx.x;
    int b  = id / 5120;
    int r  = id - b * 5120;
    int oc = r / 160;
    int od = r - oc * 160;

    __shared__ float sw[400];
    __shared__ float red[256];
    __shared__ float stat;
    for (int i = threadIdx.x; i < 400; i += 256) sw[i] = w3[oc * 400 + i];
    __syncthreads();

    int oh = threadIdx.x >> 4, ow = threadIdx.x & 15;
    float acc = b3[oc];
    for (int ic = 0; ic < 8; ++ic)
    for (int kd = 0; kd < 2; ++kd) {
        int dd = 2 * od + kd;
        const float* src = y8 + (((size_t)b * 320 + dd) * 8 + ic) * 256;
        #pragma unroll
        for (int kh = 0; kh < 5; ++kh) {
            int iy = oh + kh - 2;
            if ((unsigned)iy < 16u) {
                #pragma unroll
                for (int kw = 0; kw < 5; ++kw) {
                    int ix = ow + kw - 2;
                    if ((unsigned)ix < 16u)
                        acc += sw[((ic * 2 + kd) * 5 + kh) * 5 + kw] * src[iy * 16 + ix];
                }
            }
        }
    }
    float yv = (acc - mu3[oc]) * rsqrtf(va3[oc] + 1e-5f) * g3[oc] + be3[oc];
    float sg = sigm_(yv);

    red[threadIdx.x] = sg;
    __syncthreads();
    if (threadIdx.x == 0) { float s = 0.0f; for (int i = 0; i < 256; ++i) s += red[i]; stat = s; }
    __syncthreads();
    float mask = sg / stat * 128.0f;
    __syncthreads();
    red[threadIdx.x] = yv * mask;
    __syncthreads();
    if (threadIdx.x == 0) {
        float s = 0.0f;
        for (int i = 0; i < 256; ++i) s += red[i];
        T[((size_t)b * 160 + od) * 32 + oc] = s * (1.0f / 256.0f);
    }
}

// =====================================================================
// Tower WMMA GEMM: weights staged zero-padded (f16) into LDS so both
// fragment builds are unconditional; A is f32 row-major in global (L2).
// =====================================================================
__device__ __forceinline__ v8f wmma_tile_l(const float* __restrict__ A, int lda, int K,
                                           const _Float16* __restrict__ WH,
                                           int m0, int n0, int lane) {
    v8f acc = {};
    int am  = m0 + (lane & 15);
    int akb = (lane < 16) ? 0 : 8;
    int bn  = n0 + (lane & 15);
    int bkb = (lane < 16) ? 0 : 8;
    const _Float16* wrow = WH + bn * K;
    for (int k0 = 0; k0 < K; k0 += 32) {
        v16h a, bb;
        #pragma unroll
        for (int e = 0; e < 8; ++e) {
            a[e]     = (_Float16)A[am * lda + k0 + akb + e];
            a[e + 8] = (_Float16)A[am * lda + k0 + 16 + akb + e];
        }
        #pragma unroll
        for (int e = 0; e < 16; ++e) bb[e] = wrow[k0 + bkb + (e & 7) + ((e >> 3) << 4)];
        acc = __builtin_amdgcn_wmma_f32_16x16x32_f16(false, a, false, bb, (short)0, acc,
                                                     false, false);
    }
    return acc;
}

// mode: 0 = store(+bias), 1 = store gelu(+bias), 2 = accumulate(+bias)
__device__ __forceinline__ void gemm_stage(const float* A, int lda, int K,
                                           const float* W, int ldw, int N, int Nt,
                                           _Float16* WH, float* C, int ldc,
                                           const float* bias, int mode,
                                           int tid, int wave, int lane) {
    int Np = Nt * 16;
    for (int i = tid; i < Np * K; i += 1024) {
        int n = i / K, k = i - n * K;
        WH[i] = (n < N) ? (_Float16)W[n * ldw + k] : (_Float16)0.0f;
    }
    __syncthreads();
    int ntiles = 20 * Nt;                       // M = 320 tokens -> 20 tiles
    for (int t = wave; t < ntiles; t += 32) {   // wave-uniform: EXEC all-ones at WMMA
        int mt = t / Nt, nt = t - mt * Nt;
        int m0 = mt * 16, n0 = nt * 16;
        v8f acc = wmma_tile_l(A, lda, K, WH, m0, n0, lane);
        int nn = n0 + (lane & 15);
        if (nn < N) {
            float bv = bias ? bias[nn] : 0.0f;
            #pragma unroll
            for (int r2 = 0; r2 < 8; ++r2) {
                int mm = m0 + r2 + ((lane < 16) ? 0 : 8);
                float v = acc[r2] + bv;
                if (mode == 1) v = gelu_(v);
                if (mode == 2) C[mm * ldc + nn] += v;
                else           C[mm * ldc + nn] = v;
            }
        }
    }
}

// =====================================================================
// The Mamba tower: one persistent workgroup (32 wave32 waves) runs all 24
// blocks; GEMMs on WMMA, scan/elementwise on VALU, stages via syncthreads.
// =====================================================================
__global__ void __launch_bounds__(1024) tower_kernel(PTab tb, float* __restrict__ wsf) {
    float* T   = wsf + T_OFF;
    float* H   = wsf + H_OFF;
    float* U   = wsf + U_OFF;
    float* XZ  = wsf + XZ_OFF;
    float* XC  = wsf + XC_OFF;
    float* DBC = wsf + DBC_OFF;
    float* DTb = wsf + DT_OFF;
    float* YSb = wsf + YS_OFF;
    float* FFb = wsf + FF_OFF;

    __shared__ _Float16 WH[4096];   // largest staged weight: in_proj 128x32

    int tid  = threadIdx.x;
    int wave = tid >> 5;
    int lane = tid & 31;

    for (int blk = 0; blk < 24; ++blk) {
        const float* const* bp = &tb.p[TI_BLK + blk * 17];
        const float* n1g = bp[0];  const float* n1b = bp[1];
        const float* mng = bp[2];  const float* mnb = bp[3];
        const float* inproj = bp[4];
        const float* convw = bp[5]; const float* convb = bp[6];
        const float* xproj = bp[7];
        const float* dtw = bp[8];  const float* dtb = bp[9];
        const float* alog = bp[10]; const float* dpv = bp[11];
        const float* outproj = bp[12];
        const float* fc1w = bp[13]; const float* fc1b = bp[14];
        const float* fc2w = bp[15]; const float* fc2b = bp[16];

        // ---- LN1 -> H ; LN2 -> U ----
        if (tid < 320) {
            const float* tr = T + tid * 32;
            float mu = 0.0f;
            for (int j = 0; j < 32; ++j) mu += tr[j];
            mu *= (1.0f / 32.0f);
            float var = 0.0f;
            for (int j = 0; j < 32; ++j) { float d = tr[j] - mu; var += d * d; }
            float inv = rsqrtf(var * (1.0f / 32.0f) + 1e-5f);
            float hv[32];
            float mu2 = 0.0f;
            for (int j = 0; j < 32; ++j) {
                hv[j] = (tr[j] - mu) * inv * n1g[j] + n1b[j];
                H[tid * 32 + j] = hv[j];
                mu2 += hv[j];
            }
            mu2 *= (1.0f / 32.0f);
            float var2 = 0.0f;
            for (int j = 0; j < 32; ++j) { float d = hv[j] - mu2; var2 += d * d; }
            float inv2 = rsqrtf(var2 * (1.0f / 32.0f) + 1e-5f);
            for (int j = 0; j < 32; ++j)
                U[tid * 32 + j] = (hv[j] - mu2) * inv2 * mng[j] + mnb[j];
        }
        __threadfence(); __syncthreads();

        // ---- in_proj: (320x32)@(32x128)^T -> XZ (xs | z) ----
        gemm_stage(U, 32, 32, inproj, 32, 128, 8, WH, XZ, 128, nullptr, 0, tid, wave, lane);
        __threadfence(); __syncthreads();

        // ---- depthwise causal conv (k=4) + SiLU -> XC ----
        for (int i = tid; i < 20480; i += 1024) {
            int e   = i & 63;
            int tok = i >> 6;
            int l   = (tok < 160) ? tok : tok - 160;
            int t0  = tok - l;
            float acc = convb[e];
            #pragma unroll
            for (int k = 0; k < 4; ++k) {
                int ll = l - 3 + k;
                if (ll >= 0) acc += convw[e * 4 + k] * XZ[(size_t)(t0 + ll) * 128 + e];
            }
            XC[i] = silu_(acc);
        }
        __threadfence(); __syncthreads();

        // ---- x_proj: (320x64)@(64x34)^T -> DBC [dt0 dt1 | B(16) | C(16)] ----
        gemm_stage(XC, 64, 64, xproj, 64, 34, 3, WH, DBC, 48, nullptr, 0, tid, wave, lane);
        __threadfence(); __syncthreads();

        // ---- dt = softplus(dbc[:, :2] @ dt_w^T + dt_b) ----
        for (int i = tid; i < 20480; i += 1024) {
            int e = i & 63, tok = i >> 6;
            float v = DBC[tok * 48 + 0] * dtw[2 * e] + DBC[tok * 48 + 1] * dtw[2 * e + 1] + dtb[e];
            DTb[i] = splus_(v);
        }
        __threadfence(); __syncthreads();

        // ---- selective scan over L=160 (thread per (b,e)) ----
        if (tid < 128) {
            int bb = tid >> 6, e = tid & 63;
            float Ae[16];
            #pragma unroll
            for (int n = 0; n < 16; ++n) Ae[n] = -__expf(alog[e * 16 + n]);
            float hs[16];
            #pragma unroll
            for (int n = 0; n < 16; ++n) hs[n] = 0.0f;
            for (int l = 0; l < 160; ++l) {
                int tok = bb * 160 + l;
                float dtv = DTb[tok * 64 + e];
                float xcv = XC[tok * 64 + e];
                const float* row = DBC + tok * 48;
                float y = 0.0f;
                #pragma unroll
                for (int n = 0; n < 16; ++n) {
                    float dA = __expf(dtv * Ae[n]);
                    hs[n] = dA * hs[n] + dtv * row[2 + n] * xcv;
                    y += hs[n] * row[18 + n];
                }
                YSb[tok * 64 + e] = y;
            }
        }
        __threadfence(); __syncthreads();

        // ---- y = (ys + xc*Dp) * silu(z) ----
        for (int i = tid; i < 20480; i += 1024) {
            int e = i & 63, tok = i >> 6;
            float zv = XZ[(size_t)tok * 128 + 64 + e];
            YSb[i] = (YSb[i] + XC[i] * dpv[e]) * silu_(zv);
        }
        __threadfence(); __syncthreads();

        // ---- out_proj: accumulate residual into T ----
        gemm_stage(YSb, 64, 64, outproj, 64, 32, 2, WH, T, 32, nullptr, 2, tid, wave, lane);
        __threadfence(); __syncthreads();

        // ---- MLP: gelu(H @ fc1^T + b1) -> FF ; T += FF @ fc2^T + b2 ----
        gemm_stage(H, 32, 32, fc1w, 32, 64, 4, WH, FFb, 64, fc1b, 1, tid, wave, lane);
        __threadfence(); __syncthreads();
        gemm_stage(FFb, 64, 64, fc2w, 64, 32, 2, WH, T, 32, fc2b, 2, tid, wave, lane);
        __threadfence(); __syncthreads();
    }
}

// =====================================================================
// Head: out[b, 2j+r] = sum_c lw[c]*T[b,j,c] + lb, standardized (ddof=1)
// =====================================================================
__global__ void head_kernel(PTab tb, const float* __restrict__ wsf, float* __restrict__ out) {
    const float* T  = wsf + T_OFF;
    const float* lw = tb.p[TI_LASTW];
    const float* lb = tb.p[TI_LASTB];
    __shared__ float red[320];
    __shared__ float stats[2];
    int b = blockIdx.x, tid = threadIdx.x;
    int j = tid >> 1;
    const float* tr = T + ((size_t)b * 160 + j) * 32;
    float val = lb[0];
    for (int c = 0; c < 32; ++c) val += lw[c] * tr[c];
    red[tid] = val;
    __syncthreads();
    if (tid == 0) { float s = 0.0f; for (int i = 0; i < 320; ++i) s += red[i]; stats[0] = s * (1.0f / 320.0f); }
    __syncthreads();
    float mu = stats[0];
    red[tid] = (val - mu) * (val - mu);
    __syncthreads();
    if (tid == 0) { float s = 0.0f; for (int i = 0; i < 320; ++i) s += red[i]; stats[1] = sqrtf(s * (1.0f / 319.0f)); }
    __syncthreads();
    out[b * 320 + tid] = (val - mu) / (stats[1] + 1e-5f);
}

// =====================================================================
extern "C" void kernel_launch(void* const* d_in, const int* in_sizes, int n_in,
                              void* d_out, int out_size, void* d_ws, size_t ws_size,
                              hipStream_t stream) {
    (void)out_size; (void)ws_size;
    PTab tb;
    auto P = [&](int i) { return (const float*)d_in[i]; };
    bool insertion = (n_in > 0 && in_sizes[0] > 1000000);  // x first => insertion-order flatten
    if (insertion) {
        tb.p[TI_X] = P(0);
        for (int s = 0; s < 5; ++s)
            for (int j = 0; j < 6; ++j) tb.p[TI_S11 + 6 * s + j] = P(1 + 6 * s + j);
        for (int b = 0; b < 24; ++b)
            for (int j = 0; j < 17; ++j) tb.p[TI_BLK + 17 * b + j] = P(31 + 17 * b + j);
        tb.p[TI_LASTW] = P(439);
        tb.p[TI_LASTB] = P(440);
    } else {
        // sorted-key pytree flatten: blocks first, x last
        for (int b = 0; b < 24; ++b) {
            int s = 17 * b, c = TI_BLK + 17 * b;
            tb.p[c + 0] = P(s + 14); tb.p[c + 1] = P(s + 13);  // norm1 g,b
            tb.p[c + 2] = P(s + 12); tb.p[c + 3] = P(s + 11);  // mnorm g,b
            tb.p[c + 4] = P(s + 10);                            // in_proj
            tb.p[c + 5] = P(s + 3);  tb.p[c + 6] = P(s + 2);   // conv_w, conv_b
            tb.p[c + 7] = P(s + 16);                            // x_proj
            tb.p[c + 8] = P(s + 5);  tb.p[c + 9] = P(s + 4);   // dt_w, dt_b
            tb.p[c + 10] = P(s + 0); tb.p[c + 11] = P(s + 1);  // A_log, Dp
            tb.p[c + 12] = P(s + 15);                           // out_proj
            tb.p[c + 13] = P(s + 7); tb.p[c + 14] = P(s + 6);  // fc1_w, fc1_b
            tb.p[c + 15] = P(s + 9); tb.p[c + 16] = P(s + 8);  // fc2_w, fc2_b
        }
        tb.p[TI_LASTB] = P(408);
        tb.p[TI_LASTW] = P(409);
        const int sb[5] = {410, 416, 422, 428, 434};
        for (int s = 0; s < 5; ++s) {
            int d0 = TI_S11 + 6 * s;
            tb.p[d0 + 0] = P(sb[s] + 5);  // w
            tb.p[d0 + 1] = P(sb[s] + 0);  // b
            tb.p[d0 + 2] = P(sb[s] + 2);  // bn gamma
            tb.p[d0 + 3] = P(sb[s] + 1);  // bn beta
            tb.p[d0 + 4] = P(sb[s] + 3);  // bn mean
            tb.p[d0 + 5] = P(sb[s] + 4);  // bn var
        }
        tb.p[TI_X] = P(440);
    }

    float* wsf = (float*)d_ws;
    stem1_kernel<<<2560, 256, 0, stream>>>(tb, wsf);
    stem2_kernel<<<640, 256, 0, stream>>>(tb, wsf);
    stem3_kernel<<<10240, 256, 0, stream>>>(tb, wsf);
    tower_kernel<<<1, 1024, 0, stream>>>(tb, wsf);
    head_kernel<<<2, 320, 0, stream>>>(tb, wsf, (float*)d_out);
}
